// V2VNetFusion_661424963685
// MI455X (gfx1250) — matrix operand order repaired and checked by reading, not compiled, not run on hardware
//
#include <hip/hip_runtime.h>
#include <hip/hip_bf16.h>

// ---------------------------------------------------------------------------
// V2VNet fusion for MI455X (gfx1250, wave32, WMMA).
// Convs lowered to implicit GEMM on v_wmma_f32_16x16x32_bf16 (f32 accum),
// activations NHWC-bf16 so every B fragment is one contiguous 32B load;
// weights pre-packed in per-lane fragment order so every A fragment is one
// contiguous 32B load. SAME-padding via a zero-page pointer select.
// sched_group_barrier pins a [loads | wmma] software pipeline so B-fragment
// loads overlap matrix-core execution (partial loadcnt waits, not full drains).
// Final MLP uses v_wmma_f32_16x16x4_f32 (fp32 path).
// Constants: B=2, L=6, C=128, H=W=64, NUM_ITER=2, RATIO*DOWNSAMPLE=0.8.
// ---------------------------------------------------------------------------

#define BB 2
#define LL 6
#define CC 128
#define HH 64
#define WW 64
#define HW 4096           // 64*64
#define NINST 12          // B*L node instances

#if defined(__has_builtin)
#if __has_builtin(__builtin_amdgcn_sched_group_barrier)
#define SGB(mask, size, id) __builtin_amdgcn_sched_group_barrier(mask, size, id)
#endif
#endif
#ifndef SGB
#define SGB(mask, size, id)
#endif
// masks: 0x020 = VMEM read, 0x008 = MFMA/WMMA

typedef __attribute__((ext_vector_type(16))) __bf16 v16bf;
typedef __attribute__((ext_vector_type(8)))  float  v8f;
typedef __attribute__((ext_vector_type(2)))  float  v2f;
typedef __attribute__((ext_vector_type(8)))  unsigned int v8u;
typedef __attribute__((ext_vector_type(4)))  unsigned int v4u;

union FragBF { v16bf v; v8u u; };
union FragF2 { v2f   v; float f[2]; };

__device__ __forceinline__ unsigned short f2bf(float f) {
  unsigned int u = __float_as_uint(f);
  unsigned int r = u + 0x7FFFu + ((u >> 16) & 1u);   // round-to-nearest-even
  return (unsigned short)(r >> 16);
}
__device__ __forceinline__ float bf2f(unsigned short h) {
  return __uint_as_float(((unsigned int)h) << 16);
}
__device__ __forceinline__ void inv23(float a, float b, float e,
                                      float c, float d, float f, float* o) {
  // inverse of affine [[a b e],[c d f],[0 0 1]], first two rows
  float id = 1.0f / (a * d - b * c);
  o[0] =  d * id; o[1] = -b * id; o[2] = (b * f - d * e) * id;
  o[3] = -c * id; o[4] =  a * id; o[5] = (c * e - a * f) * id;
}

// --- per-pair warp matrices + zero-page init --------------------------------
__global__ void V2V_theta_kernel(const float* __restrict__ T4,
                                 float* __restrict__ theta_bil,
                                 float* __restrict__ theta_roi,
                                 unsigned short* __restrict__ zrow) {
  int idx = threadIdx.x;
  if (idx < CC) zrow[idx] = 0;                     // SAME-padding zero page
  if (idx >= BB * LL * LL) return;
  int b = idx / (LL * LL), rem = idx % (LL * LL);
  int i = rem / LL, j = rem % LL;
  // tm[b][j][i]: rows {0,1}, cols {0,1,3}; translation scaled by 1/0.8
  const float* Tm = T4 + (((size_t)b * LL + j) * LL + i) * 16;
  float a = Tm[0], bcoef = Tm[1], e = Tm[3] * 1.25f;
  float c = Tm[4], d     = Tm[5], f = Tm[7] * 1.25f;
  inv23(a, bcoef, e, c, d, f, theta_roi + (size_t)idx * 6);
  // get_transformation_matrix: t = center - R*center + M[:,2], center=(32,32)
  float te = 32.f - (a * 32.f + bcoef * 32.f) + e;
  float tf = 32.f - (c * 32.f + d * 32.f) + f;
  inv23(a, bcoef, te, c, d, tf, theta_bil + (size_t)idx * 6);
}

// --- pack conv weights to bf16 in exact A-fragment order:
//     [tap(9)][kc(8)][mtile(8)][lane(32)][e(16)]  (32B per lane fragment) ---
__global__ void V2V_pack_w_kernel(const float* __restrict__ msgW,
                                  const float* __restrict__ gatesW,
                                  const float* __restrict__ canW,
                                  unsigned short* __restrict__ msgWp,
                                  unsigned short* __restrict__ gWp,
                                  unsigned short* __restrict__ cWp) {
  int idx = blockIdx.x * 256 + threadIdx.x;
  const int TOT = 9 * 8 * 8 * 32 * 16;     // 294912
  if (idx >= TOT) return;
  int e    = idx & 15;
  int lane = (idx >> 4) & 31;
  int mt   = (idx >> 9) & 7;
  int kc   = (idx >> 12) & 7;
  int tap  = idx >> 15;
  int v = e >> 1;   // A 16x32 bf16 lane layout: contiguous k = 16*(lane/16) + e (matches B)
  int ka = e + ((lane >= 16) ? 16 : 0);
  int m   = mt * 16 + (lane & 15);
  int cin = kc * 32 + ka;
  msgWp[idx] = f2bf(msgW[((size_t)m * 256 + cin) * 9 + tap]);            // (C,2C,3,3)
  gWp[idx]   = f2bf(gatesW[((size_t)(CC + m) * 384 + cin) * 9 + tap]);   // update rows
  cWp[idx]   = f2bf(canW[((size_t)m * 384 + cin) * 9 + tap]);            // (C,3C,3,3)
}

// --- xf = flip/transpose of feats, stored NHWC bf16: xf[p][c] ---
__global__ void V2V_flip_tr_kernel(const float* __restrict__ feats,
                                   unsigned short* __restrict__ xf_nhwc) {
  size_t gid = (size_t)blockIdx.x * 256 + threadIdx.x;   // NINST*HW*CC
  int c = (int)(gid & (CC - 1));
  int p = (int)((gid >> 7) & (HW - 1));
  int inst = (int)(gid >> 19);
  int xx = p & 63, yy = p >> 6;
  // xf[y][x] = feats[63-x][y]
  xf_nhwc[gid] = f2bf(feats[((size_t)inst * CC + c) * HW + (size_t)(63 - xx) * 64 + yy]);
}

// --- per (node,channel) std with ddof=1 ---
__global__ void V2V_std_kernel(const float* __restrict__ feats,
                               float* __restrict__ stdv) {
  __shared__ float s1[256];
  __shared__ float s2[256];
  int nc = blockIdx.x, tid = threadIdx.x;
  const float* p = feats + (size_t)nc * HW;
  float a = 0.f, b = 0.f;
  for (int k = tid; k < HW; k += 256) { float v = p[k]; a += v; b += v * v; }
  s1[tid] = a; s2[tid] = b; __syncthreads();
  for (int st = 128; st > 0; st >>= 1) {
    if (tid < st) { s1[tid] += s1[tid + st]; s2[tid] += s2[tid + st]; }
    __syncthreads();
  }
  if (tid == 0) {
    float mean = s1[0] * (1.0f / 4096.0f);
    float var  = (s2[0] - 4096.0f * mean * mean) * (1.0f / 4095.0f);
    stdv[nc] = sqrtf(fmaxf(var, 0.f));
  }
}

// --- top-C/2 mask per node (rank by std, ties to lower index, like top_k) ---
__global__ void V2V_mask_kernel(const float* __restrict__ stdv,
                                int* __restrict__ mask, int iter) {
  __shared__ float s[CC];
  int nid = blockIdx.x, c = threadIdx.x;
  float v = stdv[nid * CC + c];
  s[c] = v; __syncthreads();
  int rank = 0;
  for (int k = 0; k < CC; ++k) {
    float u = s[k];
    rank += (u > v) || (u == v && k < c);
  }
  int m = (rank < CC / 2) ? 1 : 0;
  if (iter == 0 && nid == 0) m = 1;   // 'first' flag: mask.at[0].set(True)
  mask[nid * CC + c] = m;
}

// --- masked bilinear warp of neighbors into nb_nhwc[(inst*6+j)][p][c] ---
__device__ __forceinline__ float sampN(const unsigned short* src, int ix, int iy) {
  if (ix < 0 || ix >= WW || iy < 0 || iy >= HH) return 0.f;
  return bf2f(src[(size_t)(iy * WW + ix) * CC]);
}
__global__ void V2V_warp_nb_kernel(const unsigned short* __restrict__ xf_nhwc,
                                   const int* __restrict__ mask,
                                   const float* __restrict__ theta_bil,
                                   unsigned short* __restrict__ nb_nhwc) {
  size_t gid = (size_t)blockIdx.x * 256 + threadIdx.x;   // 72*HW*CC
  int c = (int)(gid & (CC - 1));
  int p = (int)((gid >> 7) & (HW - 1));
  int ij = (int)(gid >> 19);          // inst*6 + j
  int inst = ij / LL, j = ij % LL;
  int b = inst / LL;
  unsigned short outv;
  if (mask[inst * CC + c]) {
    const float* th = theta_bil + (size_t)ij * 6;
    int xd = p & 63, yd = p >> 6;
    float px = th[0] * xd + th[1] * yd + th[2];
    float py = th[3] * xd + th[4] * yd + th[5];
    float x0 = floorf(px), y0 = floorf(py);
    float fx = px - x0, fy = py - y0;
    int ix = (int)x0, iy = (int)y0;
    const unsigned short* src = xf_nhwc + (size_t)(b * LL + j) * HW * CC + c;
    float v00 = sampN(src, ix, iy),     v01 = sampN(src, ix + 1, iy);
    float v10 = sampN(src, ix, iy + 1), v11 = sampN(src, ix + 1, iy + 1);
    float top = v00 * (1.f - fx) + v01 * fx;
    float bot = v10 * (1.f - fx) + v11 * fx;
    outv = f2bf(top * (1.f - fy) + bot * fy);
  } else {
    outv = xf_nhwc[((size_t)inst * HW + p) * CC + c];   // ego substitute
  }
  nb_nhwc[gid] = outv;
}

// --- msg conv (implicit GEMM, bf16 WMMA) + ROI + mean over neighbors.
// block = 256 = 8 waves; wave w owns cout tile [16w,16w+16) x 16 pixels.
// Ego half of the input (channels 128..255) is identical for all neighbors:
// computed once into acc_ego and distributed (ROI scaling is linear).
__global__ void V2V_msg_agg_kernel(const unsigned short* __restrict__ nb_nhwc,
                                   const unsigned short* __restrict__ xf_nhwc,
                                   const unsigned short* __restrict__ msgWp,
                                   const float* __restrict__ msg_b,
                                   const float* __restrict__ theta_roi,
                                   const unsigned short* __restrict__ zrow,
                                   unsigned short* __restrict__ agg_nhwc) {
  int inst = blockIdx.y;
  int t = blockIdx.x;                 // 256 pixel tiles
  int y = t >> 2, xb = (t & 3) << 4;
  int lane = threadIdx.x & 31, wv = threadIdx.x >> 5;
  int mbase = wv << 4, hi = lane >> 4, l16 = lane & 15;
  int xn = xb + l16;
  const unsigned short* ego = xf_nhwc + (size_t)inst * HW * CC;
  const unsigned short* nbb = nb_nhwc + (size_t)(inst * LL) * HW * CC;
  v8f zf = {};
  v8f accj[LL];
  #pragma unroll
  for (int j = 0; j < LL; ++j) accj[j] = zf;
  v8f acce = zf;
  for (int tap = 0; tap < 9; ++tap) {
    int ys = y + tap / 3 - 1;
    int xs = xn + tap % 3 - 1;
    bool cv = (ys >= 0) && (ys < HH) && (xs >= 0) && (xs < WW);
    int po = (ys * WW + xs) * CC;               // may be junk when !cv
    const unsigned short* rowe = cv ? (ego + po) : zrow;
    const unsigned short* rowj[LL];
    #pragma unroll
    for (int j = 0; j < LL; ++j)
      rowj[j] = cv ? (nbb + (size_t)j * HW * CC + po) : zrow;
    const unsigned short* wtap = msgWp + ((((size_t)tap * 8) * 8 + wv) * 32 + lane) * 16;
    // ---- ego half (channels 128..255), shared by all neighbors ----
    {
      FragBF Ae[4], Be[4];
      #pragma unroll
      for (int kc = 0; kc < 4; ++kc) {
        Ae[kc].u = *(const v8u*)(wtap + (size_t)(kc + 4) * 8 * 32 * 16);
        Be[kc].u = *(const v8u*)(rowe + kc * 32 + hi * 16);
      }
      #pragma unroll
      for (int kc = 0; kc < 4; ++kc)
        acce = __builtin_amdgcn_wmma_f32_16x16x32_bf16(
            false, Ae[kc].v, false, Be[kc].v, (short)0, acce, false, false);
      // pipeline: [A+B loads (4 x b128)] then one WMMA, x4
      #pragma unroll
      for (int q = 0; q < 4; ++q) { SGB(0x020, 4, 0); SGB(0x008, 1, 0); }
    }
    // ---- neighbor half (channels 0..127) ----
    #pragma unroll
    for (int kc = 0; kc < 4; ++kc) {
      int coff = kc * 32 + hi * 16;             // B 32x16: k striped in lane halves
      FragBF A;
      A.u = *(const v8u*)(wtap + (size_t)kc * 8 * 32 * 16);
      FragBF Bl[LL];
      #pragma unroll
      for (int j = 0; j < LL; ++j)
        Bl[j].u = *(const v8u*)(rowj[j] + coff);
      #pragma unroll
      for (int j = 0; j < LL; ++j)
        accj[j] = __builtin_amdgcn_wmma_f32_16x16x32_bf16(
            false, A.v, false, Bl[j].v, (short)0, accj[j], false, false);
      // pipeline: A (2 x b128), then 6 x [B loads (2 x b128), WMMA]
      SGB(0x020, 2, 0);
      #pragma unroll
      for (int q = 0; q < LL; ++q) { SGB(0x020, 2, 0); SGB(0x008, 1, 0); }
    }
  }
  v8f bias = *(const v8f*)(msg_b + mbase + hi * 8);
  v8f acc = zf;
  #pragma unroll
  for (int j = 0; j < LL; ++j) {
    // roi: nearest-warp validity of ones under tm[b,j,i]
    const float* th = theta_roi + (size_t)(inst * LL + j) * 6;
    float px = rintf(th[0] * xn + th[1] * y + th[2]);
    float py = rintf(th[3] * xn + th[4] * y + th[5]);
    float roi = (px >= 0.f && px <= 63.f && py >= 0.f && py <= 63.f) ? 1.f : 0.f;
    #pragma unroll
    for (int r = 0; r < 8; ++r) acc[r] += (accj[j][r] + acce[r] + bias[r]) * roi;
  }
  // store mean as packed bf16, NHWC (16B per lane, contiguous in m)
  unsigned short* orow = agg_nhwc + ((size_t)inst * HW + y * WW + xn) * CC + mbase + hi * 8;
  v4u pk;
  #pragma unroll
  for (int q = 0; q < 4; ++q) {
    unsigned lo = f2bf(acc[2 * q]     * (1.0f / 6.0f));
    unsigned hs = f2bf(acc[2 * q + 1] * (1.0f / 6.0f));
    pk[q] = lo | (hs << 16);
  }
  *(v4u*)orow = pk;
}

// --- gates/can conv on cat=[ego;agg] (h==0 -> last C input channels dropped).
// phase 0: upd = sigmoid(conv+b). phase 1: g = upd*tanh(conv+b), scattered with
// the same flip/transpose as xf into feats_next (NCHW for next iter / MLP).
__global__ void V2V_gru_conv_kernel(const unsigned short* __restrict__ xf_nhwc,
                                    const unsigned short* __restrict__ agg_nhwc,
                                    const unsigned short* __restrict__ Wp,
                                    const float* __restrict__ bvec, int bias_off,
                                    const unsigned short* __restrict__ zrow,
                                    float* __restrict__ upd,
                                    float* __restrict__ feats_next, int phase) {
  int inst = blockIdx.y;
  int t = blockIdx.x;
  int y = t >> 2, xb = (t & 3) << 4;
  int lane = threadIdx.x & 31, wv = threadIdx.x >> 5;
  int mbase = wv << 4, hi = lane >> 4, l16 = lane & 15;
  int xn = xb + l16;
  const unsigned short* ego = xf_nhwc + (size_t)inst * HW * CC;
  const unsigned short* agp = agg_nhwc + (size_t)inst * HW * CC;
  v8f acc = {};
  for (int tap = 0; tap < 9; ++tap) {
    int ys = y + tap / 3 - 1;
    int xs = xn + tap % 3 - 1;
    bool cv = (ys >= 0) && (ys < HH) && (xs >= 0) && (xs < WW);
    int po = (ys * WW + xs) * CC;
    const unsigned short* rowe = cv ? (ego + po) : zrow;
    const unsigned short* rowa = cv ? (agp + po) : zrow;
    const unsigned short* wtap = Wp + ((((size_t)tap * 8) * 8 + wv) * 32 + lane) * 16;
    // batch-load all 8 fragment pairs of this tap, then chain 8 WMMAs
    FragBF As[8], Bs[8];
    #pragma unroll
    for (int kc = 0; kc < 8; ++kc) {
      As[kc].u = *(const v8u*)(wtap + (size_t)kc * 8 * 32 * 16);
      Bs[kc].u = (kc < 4)
          ? *(const v8u*)(rowe + kc * 32 + hi * 16)
          : *(const v8u*)(rowa + (kc - 4) * 32 + hi * 16);
    }
    #pragma unroll
    for (int kc = 0; kc < 8; ++kc)
      acc = __builtin_amdgcn_wmma_f32_16x16x32_bf16(
          false, As[kc].v, false, Bs[kc].v, (short)0, acc, false, false);
    // pipeline: 8 x [A+B loads (4 x b128), WMMA]
    #pragma unroll
    for (int q = 0; q < 8; ++q) { SGB(0x020, 4, 0); SGB(0x008, 1, 0); }
  }
  v8f bias = *(const v8f*)(bvec + bias_off + mbase + hi * 8);
  size_t ubase = ((size_t)inst * HW + y * WW + xn) * CC + mbase + hi * 8;
  if (phase == 0) {
    v8f res;
    #pragma unroll
    for (int r = 0; r < 8; ++r) res[r] = 1.0f / (1.0f + expf(-(acc[r] + bias[r])));
    *(v8f*)(upd + ubase) = res;                 // 32B contiguous store
  } else {
    v8f u = *(const v8f*)(upd + ubase);
    #pragma unroll
    for (int r = 0; r < 8; ++r) {
      int m = mbase + r + hi * 8;
      float g = u[r] * tanhf(acc[r] + bias[r]);
      // feats_next[h][w] = g[63-w][h]  (same flip/transpose as xf), NCHW
      feats_next[((size_t)inst * CC + m) * HW + (size_t)xn * WW + (63 - y)] = g;
    }
  }
}

// --- final MLP: out[b,h,w,o] = sum_c feats(node0)[c,h,w] * W[o,c] + b[o].
// fp32 WMMA 16x16x4 (full precision).
__global__ void V2V_mlp_kernel(const float* __restrict__ feats,
                               const float* __restrict__ W,
                               const float* __restrict__ bvec,
                               float* __restrict__ out) {
  int b = blockIdx.y;
  int t = blockIdx.x;
  int y = t >> 2, xb = (t & 3) << 4;
  int lane = threadIdx.x & 31, wv = threadIdx.x >> 5;
  int mbase = wv << 4, hi = lane >> 4, l16 = lane & 15;
  int xn = xb + l16;
  const float* src = feats + (size_t)(b * LL) * CC * HW;   // node 0 of batch b
  v8f acc = {};
  for (int kc = 0; kc < CC / 4; ++kc) {
    int cb = kc << 2;
    FragF2 A, Bf;
    int m = mbase + l16;
    #pragma unroll
    for (int e = 0; e < 2; ++e) {
      int k = e + (hi ? 2 : 0);    // f32 16x16x4 A/B layouts (ISA 7.12.2)
      A.f[e]  = W[(size_t)m * CC + cb + k];
      Bf.f[e] = src[(size_t)(cb + k) * HW + y * WW + xn];
    }
    acc = __builtin_amdgcn_wmma_f32_16x16x4_f32(
        false, A.v, false, Bf.v, (short)0, acc, false, false);
  }
  v8f bias = *(const v8f*)(bvec + mbase + hi * 8);
  v8f res;
  #pragma unroll
  for (int r = 0; r < 8; ++r) res[r] = acc[r] + bias[r];
  *(v8f*)(out + ((size_t)b * HW + y * WW + xn) * CC + mbase + hi * 8) = res;
}

// ---------------------------------------------------------------------------
extern "C" void kernel_launch(void* const* d_in, const int* in_sizes, int n_in,
                              void* d_out, int out_size, void* d_ws, size_t ws_size,
                              hipStream_t stream) {
  (void)in_sizes; (void)n_in; (void)out_size; (void)ws_size;
  const float* x       = (const float*)d_in[0];
  const float* T4      = (const float*)d_in[2];   // pairwise_t_matrix (B,L,L,4,4)
  const float* msgW    = (const float*)d_in[4];
  const float* msg_b   = (const float*)d_in[5];
  const float* gatesW  = (const float*)d_in[6];
  const float* gates_b = (const float*)d_in[7];
  const float* canW    = (const float*)d_in[8];
  const float* can_b   = (const float*)d_in[9];
  const float* mlpW    = (const float*)d_in[10];
  const float* mlp_b   = (const float*)d_in[11];
  float* out = (float*)d_out;

  const size_t F = (size_t)NINST * CC * HW;       // 6,291,456 elems
  char* base = (char*)d_ws;
  size_t off = 0;
  auto alloc = [&](size_t bytes) -> void* {
    void* p = base + off;
    off = (off + bytes + 255) & ~(size_t)255;
    return p;
  };
  float*          featsA   = (float*)alloc(F * 4);
  float*          featsB   = (float*)alloc(F * 4);
  unsigned short* xf_nhwc  = (unsigned short*)alloc(F * 2);
  unsigned short* nb_nhwc  = (unsigned short*)alloc(F * LL * 2);   // 72 maps
  unsigned short* agg_nhwc = (unsigned short*)alloc(F * 2);
  float*          upd      = (float*)alloc(F * 4);
  float*          stdv     = (float*)alloc((size_t)NINST * CC * 4);
  int*            mask     = (int*)alloc((size_t)NINST * CC * 4);
  float*          th_bil   = (float*)alloc((size_t)BB * LL * LL * 6 * 4);
  float*          th_roi   = (float*)alloc((size_t)BB * LL * LL * 6 * 4);
  unsigned short* msgWp    = (unsigned short*)alloc((size_t)9 * 256 * CC * 2);
  unsigned short* gWp      = (unsigned short*)alloc((size_t)9 * 256 * CC * 2);
  unsigned short* cWp      = (unsigned short*)alloc((size_t)9 * 256 * CC * 2);
  unsigned short* zrow     = (unsigned short*)alloc((size_t)CC * 2);  // zero page

  V2V_theta_kernel<<<1, 128, 0, stream>>>(T4, th_bil, th_roi, zrow);
  V2V_pack_w_kernel<<<(9 * 256 * CC + 255) / 256, 256, 0, stream>>>(
      msgW, gatesW, canW, msgWp, gWp, cWp);
  hipMemcpyAsync(featsA, x, F * 4, hipMemcpyDeviceToDevice, stream);

  float* cur = featsA;
  float* nxt = featsB;
  for (int it = 0; it < 2; ++it) {
    V2V_flip_tr_kernel<<<(unsigned)(F / 256), 256, 0, stream>>>(cur, xf_nhwc);
    V2V_std_kernel<<<NINST * CC, 256, 0, stream>>>(cur, stdv);
    V2V_mask_kernel<<<NINST, CC, 0, stream>>>(stdv, mask, it);
    V2V_warp_nb_kernel<<<(unsigned)(F * LL / 256), 256, 0, stream>>>(
        xf_nhwc, mask, th_bil, nb_nhwc);
    V2V_msg_agg_kernel<<<dim3(256, NINST), 256, 0, stream>>>(
        nb_nhwc, xf_nhwc, msgWp, msg_b, th_roi, zrow, agg_nhwc);
    V2V_gru_conv_kernel<<<dim3(256, NINST), 256, 0, stream>>>(
        xf_nhwc, agg_nhwc, gWp, gates_b, CC, zrow, upd, nullptr, 0);
    V2V_gru_conv_kernel<<<dim3(256, NINST), 256, 0, stream>>>(
        xf_nhwc, agg_nhwc, cWp, can_b, 0, zrow, upd, nxt, 1);
    float* tmp = cur; cur = nxt; nxt = tmp;
  }

  V2V_mlp_kernel<<<dim3(256, BB), 256, 0, stream>>>(cur, mlpW, mlp_b, out);
}
